// Attention_15771119911459
// MI455X (gfx1250) — matrix operand order repaired and checked by reading, compiled, not run on hardware
//
#include <hip/hip_runtime.h>
#include <hip/hip_bf16.h>

// ---------------------------------------------------------------------------
// MI455X (gfx1250) attention block: bf16 WMMA flash-attention pipeline with
// TDM (tensor_load_to_lds) staging of K/V tiles.  Wave32 throughout.
// ---------------------------------------------------------------------------

typedef __bf16 bf16;
typedef __attribute__((ext_vector_type(16))) __bf16 v16bf;
typedef __attribute__((ext_vector_type(8)))  float  v8f;
typedef unsigned int u32x4 __attribute__((ext_vector_type(4)));
typedef int i32x8 __attribute__((ext_vector_type(8)));
typedef int i32x4 __attribute__((ext_vector_type(4)));

#define HEADS 8
#define DIM_HEAD 64
#define DIM 512
#define BATCH 8
#define SEQ 2048
#define TOKENS (BATCH * SEQ)
#define SCALE 0.125f           // 64^-0.5
#define BN_EPS 1e-5f

__device__ __forceinline__ v8f wmma_bf16(v16bf a, v16bf b, v8f c) {
  // D = A(16x32 bf16) * B(32x16 bf16) + C(16x16 f32)
  return __builtin_amdgcn_wmma_f32_16x16x32_bf16(
      /*neg_a=*/false, a, /*neg_b=*/false, b,
      /*c_mod=*/(short)0, c, /*reuse_a=*/false, /*reuse_b=*/false);
}

// A operand (16x32, bf16) from LDS stored row-major [row][k] with stride rs.
// ISA: lanes 0-15 row M=lane, elems0-7=K0..7, elems8-15=K16..23;
//      lanes 16-31 same rows, elems0-7=K8..15, elems8-15=K24..31.
__device__ __forceinline__ v16bf load_a(const bf16* lds, int row0, int k0,
                                        int rs, int lane) {
  const int m = lane & 15, hf = lane >> 4;
  const bf16* base = lds + (row0 + m) * rs + k0;
  v16bf a;
#pragma unroll
  for (int j = 0; j < 8; ++j) a[j] = base[hf * 8 + j];
#pragma unroll
  for (int j = 0; j < 8; ++j) a[8 + j] = base[16 + hf * 8 + j];
  return a;
}

// B operand (32x16) where LDS holds the matrix TRANSPOSED: B[k][n]=lds[n*rs+k]
// ISA: lanes 0-15 col N=lane, elems0-7=K0..7, elems8-15=K16..23; lanes 16-31: K8..15, K24..31.
// Per-lane two runs of 8 x bf16 (16 apart)  ->  2 x ds_load_b128.
__device__ __forceinline__ v16bf load_b_nk(const bf16* lds, int k0, int n0,
                                           int rs, int lane) {
  const int n = lane & 15, hf = lane >> 4;
  const bf16* base = lds + (n0 + n) * rs + k0 + hf * 8;
  v16bf b;
#pragma unroll
  for (int j = 0; j < 8; ++j) b[j] = base[j];
#pragma unroll
  for (int j = 0; j < 8; ++j) b[8 + j] = base[16 + j];
  return b;
}

// ---------------------------------------------------------------------------
// Tensor Data Mover: 2D tile load, bf16 elements.
// D# per CDNA5 ISA:
//   group0: count=1 | lds_addr | global_addr[56:0] | type=2
//   group1: data_size=1(2B) | tensor_dim0/1 | tile_dim0/1 | dim0_stride
// groups 2/3 zero (2D tensor).  Tracked by TENSORcnt.
// This toolchain's builtin takes 6 args (extra int32x8 group before cpol).
// ---------------------------------------------------------------------------
__device__ __forceinline__ void tdm_load_2d(unsigned lds_off, const void* gptr,
                                            unsigned tensor_d0,
                                            unsigned tensor_d1,
                                            unsigned tile_d0, unsigned tile_d1,
                                            unsigned stride0) {
  const unsigned long long ga = (unsigned long long)(uintptr_t)gptr;
  u32x4 g0;
  g0[0] = 1u;                                   // count=1, no gather
  g0[1] = lds_off;                              // lds_addr (bytes)
  g0[2] = (unsigned)ga;                         // global_addr[31:0]
  g0[3] = (unsigned)((ga >> 32) & 0x1FFFFFFu) | (2u << 30);  // [56:32]|type=2
  i32x8 g1;
  g1[0] = (int)(1u << 16);                                  // data_size=2B
  g1[1] = (int)((tensor_d0 & 0xFFFFu) << 16);               // dim0 lo16
  g1[2] = (int)(((tensor_d0 >> 16) & 0xFFFFu) |
                ((tensor_d1 & 0xFFFFu) << 16));             // dim0 hi|dim1 lo
  g1[3] = (int)(((tensor_d1 >> 16) & 0xFFFFu) |
                ((tile_d0 & 0xFFFFu) << 16));               // dim1 hi|tile0
  g1[4] = (int)(tile_d1 & 0xFFFFu);                         // tile1, tile2=0
  g1[5] = (int)stride0;                                     // dim0_stride lo32
  g1[6] = 0;
  g1[7] = 0;
  const i32x4 z4 = {0, 0, 0, 0};
  const i32x8 z8 = {0, 0, 0, 0, 0, 0, 0, 0};
  __builtin_amdgcn_tensor_load_to_lds(g0, g1, z4, z4, z8, 0);
}

__device__ __forceinline__ unsigned lds_offset(const void* p) {
  return (unsigned)(uintptr_t)p;  // low 32 bits of flat addr = LDS offset
}

// ---------------------------------------------------------------------------
// Kernel 1: grouped QKV projection.  grid = (SEQ/64, BATCH*HEADS), 128 thr.
// q/k out: bf16 [b,h,p,e];  v out TRANSPOSED: bf16 [b,h,e,p] (for P*V B-op).
// ---------------------------------------------------------------------------
__global__ __launch_bounds__(128) void qkv_proj_kernel(
    const float* __restrict__ x, const float* __restrict__ wq,
    const float* __restrict__ wk, const float* __restrict__ wv,
    bf16* __restrict__ qb, bf16* __restrict__ kb, bf16* __restrict__ vb) {
  const int bh = blockIdx.y;
  const int b = bh >> 3, h = bh & 7;
  const int p0 = blockIdx.x * 64;
  const int tid = threadIdx.x, wave = tid >> 5, lane = tid & 31;

  __shared__ bf16 Xs[64 * 64];        // [row][c]
  __shared__ bf16 Ws[3][64 * 64];     // [e][c] per head (== B transposed)

  const float* xbase = x + ((size_t)b * SEQ + p0) * DIM + h * DIM_HEAD;
  for (int i = tid; i < 64 * 64; i += 128) {
    int r = i >> 6, c = i & 63;
    Xs[i] = (bf16)xbase[(size_t)r * DIM + c];
  }
  const float* wsrc[3] = {wq + h * 4096, wk + h * 4096, wv + h * 4096};
#pragma unroll
  for (int w = 0; w < 3; ++w)
    for (int i = tid; i < 4096; i += 128) Ws[w][i] = (bf16)wsrc[w][i];
  __syncthreads();

  const int row0 = wave * 16;
  const v16bf a0 = load_a(Xs, row0, 0, 64, lane);
  const v16bf a1 = load_a(Xs, row0, 32, 64, lane);
  const int hf = lane >> 4, nn = lane & 15;

  bf16* dstqk[2] = {qb, kb};
#pragma unroll
  for (int w = 0; w < 2; ++w) {
#pragma unroll
    for (int t = 0; t < 4; ++t) {
      v8f acc;
#pragma unroll
      for (int r = 0; r < 8; ++r) acc[r] = 0.0f;
      acc = wmma_bf16(a0, load_b_nk(Ws[w], 0, t * 16, 64, lane), acc);
      acc = wmma_bf16(a1, load_b_nk(Ws[w], 32, t * 16, 64, lane), acc);
      bf16* ob = dstqk[w] +
                 (((size_t)bh * SEQ) + p0 + row0 + hf * 8) * DIM_HEAD +
                 t * 16 + nn;
#pragma unroll
      for (int r = 0; r < 8; ++r) ob[(size_t)r * DIM_HEAD] = (bf16)acc[r];
    }
  }
  // V: transposed store [b,h,e,p]; 8 consecutive p per lane -> one b128 store.
#pragma unroll
  for (int t = 0; t < 4; ++t) {
    v8f acc;
#pragma unroll
    for (int r = 0; r < 8; ++r) acc[r] = 0.0f;
    acc = wmma_bf16(a0, load_b_nk(Ws[2], 0, t * 16, 64, lane), acc);
    acc = wmma_bf16(a1, load_b_nk(Ws[2], 32, t * 16, 64, lane), acc);
    union { uint4 u; bf16 e[8]; } pk;
#pragma unroll
    for (int r = 0; r < 8; ++r) pk.e[r] = (bf16)acc[r];
    const int e = t * 16 + nn;
    const size_t base = ((size_t)bh * DIM_HEAD + e) * SEQ + p0 + row0 + hf * 8;
    *(uint4*)(vb + base) = pk.u;   // 16B-aligned: p offset is multiple of 8
  }
}

// ---------------------------------------------------------------------------
// Kernel 2: flash attention.  grid = (SEQ/64, BATCH*HEADS), 128 thr (4 waves).
// K/V tiles staged by the Tensor Data Mover; online softmax per 16-row strip.
// Output: bf16 [token, h*64+e].
// ---------------------------------------------------------------------------
__global__ __launch_bounds__(128) void flash_attn_kernel(
    const bf16* __restrict__ qb, const bf16* __restrict__ kb,
    const bf16* __restrict__ vb, bf16* __restrict__ fo) {
  const int bh = blockIdx.y;
  const int b = bh >> 3, h = bh & 7;
  const int q0 = blockIdx.x * 64;
  const int tid = threadIdx.x, wave = tid >> 5, lane = tid & 31;
  const int hf = lane >> 4, nn = lane & 15;

  __shared__ bf16 Qs[64 * 64];       // [row][e]
  __shared__ bf16 Kt[32 * 64];       // [key][e]   (rs=64)
  __shared__ bf16 Vt[64 * 32];       // [e][key]   (rs=32, V stored transposed)
  __shared__ float Ps[4][16 * 32];   // per-wave P scratch (C->A relayout)

  const bf16* qsrc = qb + ((size_t)bh * SEQ + q0) * DIM_HEAD;
  for (int i = tid; i < 64 * 64; i += 128) Qs[i] = qsrc[i];
  __syncthreads();
  const v16bf qa0 = load_a(Qs, wave * 16, 0, 64, lane);
  const v16bf qa1 = load_a(Qs, wave * 16, 32, 64, lane);

  float m[8], l[8];
  v8f o[4];
#pragma unroll
  for (int r = 0; r < 8; ++r) { m[r] = -1e30f; l[r] = 0.0f; }
#pragma unroll
  for (int t = 0; t < 4; ++t)
#pragma unroll
    for (int r = 0; r < 8; ++r) o[t][r] = 0.0f;

  const bf16* ksrc = kb + (size_t)bh * SEQ * DIM_HEAD;       // [p][e]
  const bf16* vsrc = vb + (size_t)bh * DIM_HEAD * SEQ;       // [e][p]
  const unsigned kt_off = lds_offset(Kt);
  const unsigned vt_off = lds_offset(Vt);

  for (int j = 0; j < SEQ / 32; ++j) {
    __syncthreads();  // all waves done with previous K/V tiles
    if (tid == 0) {
      // K tile: 32 keys x 64 e from [p][e] tensor (row stride 64).
      tdm_load_2d(kt_off, ksrc + (size_t)j * 32 * DIM_HEAD,
                  /*tensor_d0=*/64, /*tensor_d1=*/(unsigned)(SEQ - j * 32),
                  /*tile_d0=*/64, /*tile_d1=*/32, /*stride0=*/64);
      // V tile: 64 e x 32 keys from [e][p] tensor (row stride 2048).
      tdm_load_2d(vt_off, vsrc + (size_t)j * 32,
                  /*tensor_d0=*/(unsigned)(SEQ - j * 32), /*tensor_d1=*/64,
                  /*tile_d0=*/32, /*tile_d1=*/64, /*stride0=*/SEQ);
    }
    __builtin_amdgcn_s_wait_tensorcnt(0);  // issuing wave drains TDM
    __syncthreads();                       // publish tiles to all waves

    // S[16x32] = Q_strip * K_tile^T  (B[k=e][n=key] = Kt[key][e])
    v8f s0, s1;
#pragma unroll
    for (int r = 0; r < 8; ++r) { s0[r] = 0.0f; s1[r] = 0.0f; }
    s0 = wmma_bf16(qa0, load_b_nk(Kt, 0, 0, 64, lane), s0);
    s0 = wmma_bf16(qa1, load_b_nk(Kt, 32, 0, 64, lane), s0);
    s1 = wmma_bf16(qa0, load_b_nk(Kt, 0, 16, 64, lane), s1);
    s1 = wmma_bf16(qa1, load_b_nk(Kt, 32, 16, 64, lane), s1);

    // Online softmax: rows live in 16-lane groups (C layout), reduce w/ xor.
    float alpha[8];
#pragma unroll
    for (int r = 0; r < 8; ++r) {
      float v0 = s0[r] * SCALE, v1 = s1[r] * SCALE;
      float rm = fmaxf(v0, v1);
      rm = fmaxf(rm, __shfl_xor(rm, 1));
      rm = fmaxf(rm, __shfl_xor(rm, 2));
      rm = fmaxf(rm, __shfl_xor(rm, 4));
      rm = fmaxf(rm, __shfl_xor(rm, 8));
      const float mn = fmaxf(m[r], rm);
      alpha[r] = __expf(m[r] - mn);
      m[r] = mn;
      v0 = __expf(v0 - mn);
      v1 = __expf(v1 - mn);
      float rs = v0 + v1;
      rs += __shfl_xor(rs, 1);
      rs += __shfl_xor(rs, 2);
      rs += __shfl_xor(rs, 4);
      rs += __shfl_xor(rs, 8);
      l[r] = l[r] * alpha[r] + rs;
      Ps[wave][(r + 8 * hf) * 32 + nn] = v0;        // cols 0..15
      Ps[wave][(r + 8 * hf) * 32 + 16 + nn] = v1;   // cols 16..31
    }
#pragma unroll
    for (int t = 0; t < 4; ++t)
#pragma unroll
      for (int r = 0; r < 8; ++r) o[t][r] *= alpha[r];

    // Wave-private LDS RAW: DS ops are in-order per wave; fence explicitly.
    asm volatile("s_wait_dscnt 0" ::: "memory");

    // Relayout P (f32, row-major 16x32) into WMMA A-operand bf16.
    v16bf pa;
    {
      const float* pb = &Ps[wave][(lane & 15) * 32];
#pragma unroll
      for (int jj = 0; jj < 8; ++jj) pa[jj] = (bf16)pb[hf * 8 + jj];
#pragma unroll
      for (int jj = 0; jj < 8; ++jj) pa[8 + jj] = (bf16)pb[16 + hf * 8 + jj];
    }

    // O += P * V   (B[k=key][n=e] = Vt[e][key] -> contiguous load_b_nk)
#pragma unroll
    for (int t = 0; t < 4; ++t)
      o[t] = wmma_bf16(pa, load_b_nk(Vt, 0, t * 16, 32, lane), o[t]);
  }

  // Epilogue: normalize and store bf16 [token, 512].
  const int row = q0 + wave * 16 + hf * 8;
  bf16* ob = fo + ((size_t)b * SEQ + row) * DIM + h * DIM_HEAD + nn;
#pragma unroll
  for (int t = 0; t < 4; ++t) {
#pragma unroll
    for (int r = 0; r < 8; ++r) {
      ob[(size_t)r * DIM + t * 16] = (bf16)(o[t][r] / l[r]);
    }
  }
}

// ---------------------------------------------------------------------------
// Kernel 3: output projection GEMM + fused BN statistics.
// out[bp,d] = sum_i fo[bp,i] * wo[d,i].  grid=(TOKENS/64, DIM/64), 128 thr.
// ---------------------------------------------------------------------------
__global__ __launch_bounds__(128) void out_proj_kernel(
    const bf16* __restrict__ fo, const float* __restrict__ wo,
    float* __restrict__ out, float* __restrict__ sums,
    float* __restrict__ sumsq) {
  const int row_blk = blockIdx.x * 64;
  const int d0 = blockIdx.y * 64;
  const int tid = threadIdx.x, wave = tid >> 5, lane = tid & 31;
  const int hf = lane >> 4, nn = lane & 15;

  __shared__ bf16 As[64 * 32];   // [row][k]
  __shared__ bf16 Bs[64 * 32];   // [n][k] (B transposed: wo is [d][i]=[n][k])

  v8f acc[4];
#pragma unroll
  for (int t = 0; t < 4; ++t)
#pragma unroll
    for (int r = 0; r < 8; ++r) acc[t][r] = 0.0f;

  for (int k0 = 0; k0 < DIM; k0 += 32) {
    __syncthreads();
    for (int i = tid; i < 64 * 32; i += 128) {
      int r = i >> 5, c = i & 31;
      As[i] = fo[((size_t)row_blk + r) * DIM + k0 + c];
      Bs[i] = (bf16)wo[((size_t)d0 + r) * DIM + k0 + c];  // Bs[n*32+k]
    }
    __syncthreads();
    const v16bf a = load_a(As, wave * 16, 0, 32, lane);
#pragma unroll
    for (int t = 0; t < 4; ++t)
      acc[t] = wmma_bf16(a, load_b_nk(Bs, 0, t * 16, 32, lane), acc[t]);
  }

  const int row = row_blk + wave * 16 + hf * 8;
#pragma unroll
  for (int t = 0; t < 4; ++t) {
    const int col = d0 + t * 16 + nn;
    float s = 0.0f, ss = 0.0f;
#pragma unroll
    for (int r = 0; r < 8; ++r) {
      const float v = acc[t][r];
      out[((size_t)row + r) * DIM + col] = v;
      s += v;
      ss += v * v;
    }
    atomicAdd(&sums[col], s);
    atomicAdd(&sumsq[col], ss);
  }
}

// ---------------------------------------------------------------------------
// Kernel 4: BatchNorm finalize + ReLU, in place on d_out.
// ---------------------------------------------------------------------------
__global__ __launch_bounds__(256) void bn_relu_kernel(
    float* __restrict__ out, const float* __restrict__ sums,
    const float* __restrict__ sumsq, const float* __restrict__ gamma,
    const float* __restrict__ beta) {
  const size_t idx = (size_t)blockIdx.x * 256 + threadIdx.x;
  const size_t N = (size_t)TOKENS * DIM;
  if (idx >= N) return;
  const int c = (int)(idx & (DIM - 1));
  const float inv = 1.0f / (float)TOKENS;
  const float mean = sums[c] * inv;
  const float var = sumsq[c] * inv - mean * mean;
  const float v =
      (out[idx] - mean) * rsqrtf(var + BN_EPS) * gamma[c] + beta[c];
  out[idx] = fmaxf(v, 0.0f);
}

// ---------------------------------------------------------------------------
extern "C" void kernel_launch(void* const* d_in, const int* in_sizes, int n_in,
                              void* d_out, int out_size, void* d_ws,
                              size_t ws_size, hipStream_t stream) {
  const float* x = (const float*)d_in[0];
  const float* wq = (const float*)d_in[1];
  const float* wk = (const float*)d_in[2];
  const float* wv = (const float*)d_in[3];
  const float* wo = (const float*)d_in[4];
  const float* gamma = (const float*)d_in[5];
  const float* beta = (const float*)d_in[6];
  float* out = (float*)d_out;

  // Workspace: 3x QKV (bf16) + flash output (bf16) + BN stats (f32).
  const size_t QKV_ELEMS = (size_t)BATCH * HEADS * SEQ * DIM_HEAD;  // 8.4M
  char* ws = (char*)d_ws;
  bf16* qb = (bf16*)ws;
  bf16* kb = qb + QKV_ELEMS;
  bf16* vb = kb + QKV_ELEMS;               // NOTE: [b,h,e,p] transposed layout
  bf16* fo = vb + QKV_ELEMS;               // TOKENS*DIM bf16
  float* sums = (float*)(fo + (size_t)TOKENS * DIM);
  float* sumsq = sums + DIM;

  (void)hipMemsetAsync(sums, 0, 2 * DIM * sizeof(float), stream);

  qkv_proj_kernel<<<dim3(SEQ / 64, BATCH * HEADS), 128, 0, stream>>>(
      x, wq, wk, wv, qb, kb, vb);
  flash_attn_kernel<<<dim3(SEQ / 64, BATCH * HEADS), 128, 0, stream>>>(
      qb, kb, vb, fo);
  out_proj_kernel<<<dim3(TOKENS / 64, DIM / 64), 128, 0, stream>>>(
      fo, wo, out, sums, sumsq);
  const size_t N = (size_t)TOKENS * DIM;
  bn_relu_kernel<<<(unsigned)((N + 255) / 256), 256, 0, stream>>>(
      out, sums, sumsq, gamma, beta);
}
